// MambaStateClassifier_4707284157169
// MI455X (gfx1250) — compile-verified, matrix operand order checked by reading, not run on hardware
//
#include <hip/hip_runtime.h>
#include <hip/hip_bf16.h>

#define B_ 8
#define L_ 1024
#define T_ (B_ * L_)          // 8192 tokens
#define DM 128                // d_model
#define DI 256                // d_inner
#define DS 16                 // d_state
#define NPADXP 64             // x_proj out dim padded 40 -> 64 (N % 32 == 0)
#define SCHUNK 16             // scan: timesteps staged per LDS refill

typedef __attribute__((ext_vector_type(16))) __bf16 v16bf;
typedef __attribute__((ext_vector_type(8)))  __bf16 v8bf;
typedef __attribute__((ext_vector_type(8)))  float  v8f;

// ---------------------------------------------------------------------------
// WMMA GEMM:  C[m,n] = sum_k A[m,k] * W[n,k]  (+ resid[m,n] if given)
// A: M x K row-major bf16, W: N x K row-major bf16, C: M x N f32.
// M mult of 16, N mult of 32, K mult of 32.
// One wave per 16x32 output tile: two accumulators share one A fragment
// (halves A traffic, two independent back-to-back v_wmma per K step).
// ---------------------------------------------------------------------------
__global__ void gemm_bf16_wmma(const __bf16* __restrict__ A,
                               const __bf16* __restrict__ W,
                               const float*  __restrict__ resid,
                               float* __restrict__ C,
                               int M, int N, int K) {
    int wave   = blockIdx.x * (blockDim.x >> 5) + (threadIdx.x >> 5);
    int lane   = threadIdx.x & 31;
    int tilesN = N >> 5;                    // 32-wide column tiles
    int tm = wave / tilesN;
    int tn = wave - tm * tilesN;
    if (tm * 16 >= M) return;               // wave-uniform: EXEC stays all-1s

    int hi = lane >> 4;                     // 0: lanes 0-15, 1: lanes 16-31
    int r  = lane & 15;
    const __bf16* Ap  = A + (size_t)(tm * 16 + r) * K;
    const __bf16* Wp0 = W + (size_t)(tn * 32 + r) * K;
    const __bf16* Wp1 = Wp0 + (size_t)16 * K;

    v8f acc0 = {}, acc1 = {};
    for (int k = 0; k < K; k += 32) {
        __builtin_prefetch(Ap  + k + 64, 0, 0);
        __builtin_prefetch(Wp0 + k + 64, 0, 0);
        __builtin_prefetch(Wp1 + k + 64, 0, 0);
        union { v16bf v; v8bf h[2]; } a, b0, b1;
        // 16-bit A layout: lanes 0-15 hold K {k..k+7, k+16..k+23},
        //                  lanes 16-31 hold K {k+8..k+15, k+24..k+31}
        a.h[0]  = *(const v8bf*)(Ap  + k + hi * 8);
        a.h[1]  = *(const v8bf*)(Ap  + k + 16 + hi * 8);
        b0.h[0] = *(const v8bf*)(Wp0 + k + hi * 8);
        b0.h[1] = *(const v8bf*)(Wp0 + k + 16 + hi * 8);
        b1.h[0] = *(const v8bf*)(Wp1 + k + hi * 8);
        b1.h[1] = *(const v8bf*)(Wp1 + k + 16 + hi * 8);
        acc0 = __builtin_amdgcn_wmma_f32_16x16x32_bf16(
            false, a.v, false, b0.v, (short)0, acc0, false, false);
        acc1 = __builtin_amdgcn_wmma_f32_16x16x32_bf16(
            false, a.v, false, b1.v, (short)0, acc1, false, false);
    }

    // C/D layout: VGPR e -> row (hi?8:0)+e, col = lane&15
    int n0    = tn * 32 + r;
    int mbase = tm * 16 + hi * 8;
    #pragma unroll
    for (int e = 0; e < 8; ++e) {
        size_t off0 = (size_t)(mbase + e) * N + n0;
        float v0 = acc0[e];
        float v1 = acc1[e];
        if (resid) { v0 += resid[off0]; v1 += resid[off0 + 16]; }
        C[off0]      = v0;
        C[off0 + 16] = v1;
    }
}

// --------------------------- weight conversion -----------------------------
__global__ void f32_to_bf16_kernel(const float* __restrict__ s,
                                   __bf16* __restrict__ d, int n) {
    int i = blockIdx.x * blockDim.x + threadIdx.x;
    if (i < n) d[i] = (__bf16)s[i];
}

// x_proj_w (4,40,256) -> padded bf16 (4,64,256), rows 40..63 zero
__global__ void convert_pad_xp_kernel(const float* __restrict__ s,
                                      __bf16* __restrict__ d) {
    int i = blockIdx.x * blockDim.x + threadIdx.x;     // 4*64*256
    if (i >= 4 * NPADXP * DI) return;
    int k = i & (DI - 1);
    int n = (i / DI) % NPADXP;
    int l = i / (NPADXP * DI);
    d[i] = (n < 40) ? (__bf16)s[((size_t)l * 40 + n) * DI + k] : (__bf16)0.f;
}

// ------------------------------- RMSNorm -----------------------------------
// one wave (32 lanes) per token, 4 elems/lane, wave32 shfl reduction
__global__ void rmsnorm_kernel(const float* __restrict__ h,
                               const float* __restrict__ w,
                               __bf16* __restrict__ out) {
    int tok  = blockIdx.x * 8 + (threadIdx.x >> 5);
    int lane = threadIdx.x & 31;
    const float* row = h + (size_t)tok * DM;
    float v[4], ss = 0.f;
    #pragma unroll
    for (int j = 0; j < 4; ++j) { v[j] = row[lane + 32 * j]; ss += v[j] * v[j]; }
    #pragma unroll
    for (int m = 16; m > 0; m >>= 1) ss += __shfl_xor(ss, m, 32);
    float sc = rsqrtf(ss * (1.f / DM) + 1e-5f);
    #pragma unroll
    for (int j = 0; j < 4; ++j)
        out[(size_t)tok * DM + lane + 32 * j] = (__bf16)(v[j] * sc * w[lane + 32 * j]);
}

// --------------------- depthwise causal conv (K=4) + SiLU ------------------
__global__ void conv_silu_kernel(const float* __restrict__ xz,
                                 const float* __restrict__ cw,
                                 const float* __restrict__ cb,
                                 float* __restrict__ xsf,
                                 __bf16* __restrict__ xsbf) {
    int idx = blockIdx.x * blockDim.x + threadIdx.x;   // T_*DI
    if (idx >= T_ * DI) return;
    int d   = idx & (DI - 1);
    int tok = idx >> 8;
    int t   = tok & (L_ - 1);
    size_t base = (size_t)(tok - t) * 512;             // batch start in xz
    const float* w = cw + d * 4;
    float acc = cb[d];
    #pragma unroll
    for (int j = 0; j < 4; ++j) {
        int tt = t - 3 + j;
        if (tt >= 0) acc += w[j] * xz[base + (size_t)tt * 512 + d];
    }
    float s = acc / (1.f + __expf(-acc));              // SiLU
    xsf[idx]  = s;
    xsbf[idx] = (__bf16)s;
}

// ----------------------- dt_proj (K=8) + softplus --------------------------
__global__ void dtproj_kernel(const float* __restrict__ dbc,
                              const float* __restrict__ dtw,
                              const float* __restrict__ dtb,
                              float* __restrict__ delta) {
    int idx = blockIdx.x * blockDim.x + threadIdx.x;   // T_*DI
    if (idx >= T_ * DI) return;
    int d   = idx & (DI - 1);
    int tok = idx >> 8;
    const float* dl = dbc + (size_t)tok * NPADXP;      // first 8 cols = dlt
    const float* w  = dtw + d * 8;
    float acc = dtb[d];
    #pragma unroll
    for (int r = 0; r < 8; ++r) acc += dl[r] * w[r];
    delta[idx] = (acc > 20.f) ? acc : log1pf(__expf(acc));   // softplus
}

// ------------------------------ selective scan -----------------------------
// one block per batch, one thread per channel; 16 states in registers.
// B/C vectors for SCHUNK timesteps staged per barrier pair (coalesced loads,
// broadcast LDS reads) -> 128 barriers per layer instead of 2048.
__global__ void scan_kernel(const float* __restrict__ xs,
                            const float* __restrict__ delta,
                            const float* __restrict__ dbc,
                            const float* __restrict__ A_log,
                            const float* __restrict__ Dp,
                            const float* __restrict__ xz,
                            __bf16* __restrict__ ybf) {
    int b = blockIdx.x;
    int d = threadIdx.x;                               // 0..255
    __shared__ float sB[SCHUNK][DS];
    __shared__ float sC[SCHUNK][DS];
    float A[DS], h[DS];
    #pragma unroll
    for (int n = 0; n < DS; ++n) {
        A[n] = -__expf(A_log[d * DS + n]);
        h[n] = 0.f;
    }
    float Dd = Dp[d];
    size_t base = (size_t)b * L_;
    int ti = d >> 4;                                   // staging: token-in-chunk
    int j  = d & 15;                                   // staging: state index
    for (int t0 = 0; t0 < L_; t0 += SCHUNK) {
        {
            size_t tok = base + t0 + ti;
            sB[ti][j] = dbc[tok * NPADXP +  8 + j];
            sC[ti][j] = dbc[tok * NPADXP + 24 + j];
        }
        __syncthreads();
        for (int tt = 0; tt < SCHUNK; ++tt) {
            size_t tok = base + t0 + tt;
            float dt = delta[tok * DI + d];
            float u  = xs[tok * DI + d];
            float du = dt * u;
            float y  = 0.f;
            #pragma unroll
            for (int n = 0; n < DS; ++n) {
                h[n] = __expf(dt * A[n]) * h[n] + du * sB[tt][n];
                y += h[n] * sC[tt][n];
            }
            y += u * Dd;
            float rg = xz[tok * 512 + DI + d];         // residual gate half
            y *= rg / (1.f + __expf(-rg));             // y * silu(res)
            ybf[tok * DI + d] = (__bf16)y;
        }
        __syncthreads();
    }
}

// --------------- final: rmsnorm(last tok) + batchnorm + relu + head --------
__global__ void final_head_kernel(const float* __restrict__ h,
                                  const float* __restrict__ nfw,
                                  const float* __restrict__ gamma,
                                  const float* __restrict__ beta,
                                  const float* __restrict__ hw,
                                  const float* __restrict__ hb,
                                  float* __restrict__ out) {
    __shared__ float feat[B_][DM];
    __shared__ float featn[B_][DM];
    int tid  = threadIdx.x;                            // 256
    int b    = tid >> 5;
    int lane = tid & 31;
    const float* row = h + ((size_t)b * L_ + (L_ - 1)) * DM;
    float v[4], ss = 0.f;
    #pragma unroll
    for (int j = 0; j < 4; ++j) { v[j] = row[lane + 32 * j]; ss += v[j] * v[j]; }
    #pragma unroll
    for (int m = 16; m > 0; m >>= 1) ss += __shfl_xor(ss, m, 32);
    float sc = rsqrtf(ss * (1.f / DM) + 1e-5f);
    #pragma unroll
    for (int j = 0; j < 4; ++j)
        feat[b][lane + 32 * j] = v[j] * sc * nfw[lane + 32 * j];
    __syncthreads();
    if (tid < DM) {
        float mu = 0.f;
        for (int bb = 0; bb < B_; ++bb) mu += feat[bb][tid];
        mu *= (1.f / B_);
        float var = 0.f;
        for (int bb = 0; bb < B_; ++bb) { float dd = feat[bb][tid] - mu; var += dd * dd; }
        var *= (1.f / B_);
        float s = rsqrtf(var + 1e-5f);
        for (int bb = 0; bb < B_; ++bb) {
            float f = (feat[bb][tid] - mu) * s * gamma[tid] + beta[tid];
            featn[bb][tid] = f > 0.f ? f : 0.f;
        }
    }
    __syncthreads();
    if (tid < 16) {
        int bb = tid >> 1, s = tid & 1;
        float acc = hb[s];
        for (int d = 0; d < DM; ++d) acc += featn[bb][d] * hw[s * DM + d];
        out[bb * 2 + s] = acc;
    }
}

// ===========================================================================
extern "C" void kernel_launch(void* const* d_in, const int* in_sizes, int n_in,
                              void* d_out, int out_size, void* d_ws, size_t ws_size,
                              hipStream_t stream) {
    const float* x         = (const float*)d_in[0];
    const float* in_proj_w = (const float*)d_in[1];   // (4, 512, 128)
    const float* conv_w    = (const float*)d_in[2];   // (4, 256, 1, 4)
    const float* conv_b    = (const float*)d_in[3];   // (4, 256)
    const float* x_proj_w  = (const float*)d_in[4];   // (4, 40, 256)
    const float* dt_proj_w = (const float*)d_in[5];   // (4, 256, 8)
    const float* dt_proj_b = (const float*)d_in[6];   // (4, 256)
    const float* A_log     = (const float*)d_in[7];   // (4, 256, 16)
    const float* D_param   = (const float*)d_in[8];   // (4, 256)
    const float* out_proj_w= (const float*)d_in[9];   // (4, 128, 256)
    const float* norm_w    = (const float*)d_in[10];  // (4, 128)
    const float* norm_f_w  = (const float*)d_in[11];  // (128,)
    const float* bn_gamma  = (const float*)d_in[12];
    const float* bn_beta   = (const float*)d_in[13];
    const float* head_w    = (const float*)d_in[14];  // (2, 128)
    const float* head_b    = (const float*)d_in[15];

    // ---- carve workspace ----
    char* p = (char*)d_ws;
    auto carve = [&](size_t bytes) -> void* {
        void* r = (void*)p;
        p += (bytes + 255) & ~(size_t)255;
        return r;
    };
    float*  h1      = (float*) carve((size_t)T_ * DM * 4);
    float*  h2      = (float*) carve((size_t)T_ * DM * 4);
    __bf16* xn_bf   = (__bf16*)carve((size_t)T_ * DM * 2);
    float*  xz      = (float*) carve((size_t)T_ * 512 * 4);
    float*  xs_f    = (float*) carve((size_t)T_ * DI * 4);
    __bf16* xs_bf   = (__bf16*)carve((size_t)T_ * DI * 2);
    float*  dbc     = (float*) carve((size_t)T_ * NPADXP * 4);
    float*  delta   = (float*) carve((size_t)T_ * DI * 4);
    __bf16* y_bf    = (__bf16*)carve((size_t)T_ * DI * 2);
    __bf16* w_in_bf = (__bf16*)carve((size_t)4 * 512 * DM * 2);
    __bf16* w_xp_bf = (__bf16*)carve((size_t)4 * NPADXP * DI * 2);
    __bf16* w_out_bf= (__bf16*)carve((size_t)4 * DM * DI * 2);

    // ---- one-time weight conversion (deterministic, replayed every call) ----
    {
        int n = 4 * 512 * DM;
        f32_to_bf16_kernel<<<(n + 255) / 256, 256, 0, stream>>>(in_proj_w, w_in_bf, n);
        n = 4 * DM * DI;
        f32_to_bf16_kernel<<<(n + 255) / 256, 256, 0, stream>>>(out_proj_w, w_out_bf, n);
        n = 4 * NPADXP * DI;
        convert_pad_xp_kernel<<<(n + 255) / 256, 256, 0, stream>>>(x_proj_w, w_xp_bf);
    }

    // h = x
    hipMemcpyAsync(h1, x, (size_t)T_ * DM * 4, hipMemcpyDeviceToDevice, stream);

    auto launch_gemm = [&](const __bf16* A, const __bf16* W, const float* resid,
                           float* C, int M, int N, int K) {
        int tiles  = (M / 16) * (N / 32);
        int blocks = (tiles + 7) / 8;                  // 8 waves per 256-thread block
        gemm_bf16_wmma<<<blocks, 256, 0, stream>>>(A, W, resid, C, M, N, K);
    };

    float* hcur = h1;
    float* hnxt = h2;
    for (int i = 0; i < 4; ++i) {
        // 1. rmsnorm -> bf16
        rmsnorm_kernel<<<T_ / 8, 256, 0, stream>>>(hcur, norm_w + i * DM, xn_bf);
        // 2. in_proj: (T,128) x (512,128)^T -> xz (T,512)
        launch_gemm(xn_bf, w_in_bf + (size_t)i * 512 * DM, nullptr, xz, T_, 512, DM);
        // 3. depthwise conv + SiLU
        conv_silu_kernel<<<(T_ * DI) / 256, 256, 0, stream>>>(
            xz, conv_w + (size_t)i * DI * 4, conv_b + (size_t)i * DI, xs_f, xs_bf);
        // 4. x_proj: (T,256) x (64,256)^T -> dbc (T,64)
        launch_gemm(xs_bf, w_xp_bf + (size_t)i * NPADXP * DI, nullptr, dbc, T_, NPADXP, DI);
        // 5. dt_proj + softplus
        dtproj_kernel<<<(T_ * DI) / 256, 256, 0, stream>>>(
            dbc, dt_proj_w + (size_t)i * DI * 8, dt_proj_b + (size_t)i * DI, delta);
        // 6. selective scan + gate -> y_bf
        scan_kernel<<<B_, DI, 0, stream>>>(
            xs_f, delta, dbc, A_log + (size_t)i * DI * DS, D_param + (size_t)i * DI,
            xz, y_bf);
        // 7. out_proj + residual: (T,256) x (128,256)^T + hcur -> hnxt
        launch_gemm(y_bf, w_out_bf + (size_t)i * DM * DI, hcur, hnxt, T_, DM, DI);
        float* tmp = hcur; hcur = hnxt; hnxt = tmp;
    }

    final_head_kernel<<<1, 256, 0, stream>>>(
        hcur, norm_f_w, bn_gamma, bn_beta, head_w, head_b, (float*)d_out);
}
